// BlockAutoregressiveNetwork_43224550867535
// MI455X (gfx1250) — matrix operand order repaired and checked by reading, not compiled, not run on hardware
//
#include <hip/hip_runtime.h>

#define NB    128      // n_blocks
#define BDIM  32       // block_dim
#define BATCH 256
#define HF    4096     // hidden features = NB*BDIM

typedef __attribute__((ext_vector_type(16))) __bf16       v16bf;
typedef __attribute__((ext_vector_type(8)))  float        v8f;
typedef __attribute__((ext_vector_type(4)))  unsigned int v4u;

union BfVec { v16bf v; v4u u[2]; };

__device__ __forceinline__ unsigned short f2bf(float f) {
  unsigned int u = __float_as_uint(f);
  u += 0x7fffu + ((u >> 16) & 1u);          // round-to-nearest-even
  return (unsigned short)(u >> 16);
}

__device__ __forceinline__ float wave_max(float v) {
#pragma unroll
  for (int off = 16; off; off >>= 1) v = fmaxf(v, __shfl_xor(v, off, 32));
  return v;
}
__device__ __forceinline__ float wave_sum(float v) {
#pragma unroll
  for (int off = 16; off; off >>= 1) v += __shfl_xor(v, off, 32);
  return v;
}

// ---------------------------------------------------------------------------
// x (f32) -> bf16
__global__ void k_cvt_bf16(const float* __restrict__ x,
                           unsigned short* __restrict__ o, int n) {
  int t = blockIdx.x * blockDim.x + threadIdx.x;
  if (t < n) o[t] = f2bf(x[t]);
}

// ---------------------------------------------------------------------------
// Layer0 prep: W0 (4096 x 128), bs=(32,1).  Row r, block i=r>>5:
//   col c < i : raw, c == i : exp, else 0.  Write full 128-col bf16 row
//   (zero padded) + ld0[r] = ls + W[r,i] - log||row||.
__global__ void k_prep0(const float* __restrict__ W, const float* __restrict__ ls,
                        unsigned short* __restrict__ Wbf, float* __restrict__ ld0) {
  int r = blockIdx.x * (blockDim.x >> 5) + (threadIdx.x >> 5);
  int lane = threadIdx.x & 31;
  if (r >= HF) return;
  int i = r >> 5;
  const float* w = W + (size_t)r * NB;
  float wm[4]; float ss = 0.f;
#pragma unroll
  for (int t = 0; t < 4; ++t) {
    int c = lane + 32 * t;
    float x = w[c];
    float v = (c < i) ? x : ((c == i) ? __expf(x) : 0.f);
    wm[t] = v; ss += v * v;
  }
  ss = wave_sum(ss);
  float lsv = ls[r];
  float scale = __expf(lsv) * rsqrtf(ss);
#pragma unroll
  for (int t = 0; t < 4; ++t)
    Wbf[(size_t)r * NB + lane + 32 * t] = f2bf(wm[t] * scale);
  if (lane == 0) ld0[r] = lsv + w[i] - 0.5f * __logf(ss);
}

// ---------------------------------------------------------------------------
// Layer1 prep: W1 (4096 x 4096), bs=(32,32). Row r, block i=r>>5:
//   cols [0,32i) raw, [32i,32i+32) exp, rest zero (never stored/read).
//   Writes bf16 triangle + ld1[i][r&31][c] (128x32x32).
__global__ void k_prep1(const float* __restrict__ W, const float* __restrict__ ls,
                        unsigned short* __restrict__ Wbf, float* __restrict__ ld1) {
  int r = blockIdx.x * (blockDim.x >> 5) + (threadIdx.x >> 5);
  int lane = threadIdx.x & 31;
  if (r >= HF) return;
  int i = r >> 5;
  int L = (i + 1) << 5;
  int d0 = i << 5;
  const float* w = W + (size_t)r * HF;
  float ss = 0.f;
  for (int c = lane; c < L; c += 32) {
    float x = w[c];
    float v = (c >= d0) ? __expf(x) : x;
    ss += v * v;
  }
  ss = wave_sum(ss);
  float lsv = ls[r];
  float lognorm = 0.5f * __logf(ss);
  float scale = __expf(lsv - lognorm);
  for (int c = lane; c < L; c += 32) {
    float x = w[c];
    float v = (c >= d0) ? __expf(x) : x;
    Wbf[(size_t)r * HF + c] = f2bf(v * scale);
  }
  ld1[(size_t)i * 1024 + (size_t)(r & 31) * 32 + lane] = lsv + w[d0 + lane] - lognorm;
}

// ---------------------------------------------------------------------------
// Layer2 prep: W2 (128 x 4096), bs=(1,32). Row r: cols [0,32r) raw,
//   [32r,32r+32) exp, rest zero (stored as zeros -> GEMM2 pads safely).
__global__ void k_prep2(const float* __restrict__ W, const float* __restrict__ ls,
                        unsigned short* __restrict__ Wbf, float* __restrict__ ld2) {
  int r = blockIdx.x * (blockDim.x >> 5) + (threadIdx.x >> 5);
  int lane = threadIdx.x & 31;
  if (r >= NB) return;
  int L = (r + 1) << 5;
  int d0 = r << 5;
  const float* w = W + (size_t)r * HF;
  float ss = 0.f;
  for (int c = lane; c < L; c += 32) {
    float x = w[c];
    float v = (c >= d0) ? __expf(x) : x;
    ss += v * v;
  }
  ss = wave_sum(ss);
  float lsv = ls[r];
  float lognorm = 0.5f * __logf(ss);
  float scale = __expf(lsv - lognorm);
  for (int c = lane; c < HF; c += 32) {
    float v = 0.f;
    if (c < L) { float x = w[c]; v = ((c >= d0) ? __expf(x) : x) * scale; }
    Wbf[(size_t)r * HF + c] = f2bf(v);
  }
  ld2[r * 32 + lane] = lsv + w[d0 + lane] - lognorm;
}

// ---------------------------------------------------------------------------
// Batch-independent: ys[n][j] = max_k ld1[n][k][j];  E1 = exp(ld1 - ys)
__global__ void k_prepE1(const float* __restrict__ ld1,
                         float* __restrict__ ys, float* __restrict__ E1) {
  int t = blockIdx.x * blockDim.x + threadIdx.x;   // n*32 + j
  if (t >= NB * 32) return;
  int n = t >> 5, j = t & 31;
  const float* p = ld1 + (size_t)n * 1024 + j;
  float m = -1e30f;
#pragma unroll 8
  for (int k = 0; k < 32; ++k) m = fmaxf(m, p[k * 32]);
  ys[t] = m;
#pragma unroll 8
  for (int k = 0; k < 32; ++k)
    E1[(size_t)n * 1024 + k * 32 + j] = __expf(p[k * 32] - m);
}

// ---------------------------------------------------------------------------
// bf16 WMMA GEMM:  Y[m][n] = sum_k A[m][k] * Wn[n][k] + bias[n]
// One wave computes a 32(M) x 32(N) tile: 2x2 grid of 16x16 WMMA tiles,
// 4 accumulators; each A fragment feeds 2 WMMAs and each B fragment feeds
// 2 WMMAs (8 x B128 loads per 4 x v_wmma_f32_16x16x32_bf16).
// klimMode: 0=full K, 1=klim=32*((n0>>5)+1) (layer1 triangle),
//           2=klim=min(K,32*(n0+32)) (layer2, zero padded).
// epi: 0 -> tanh activation + tanh log-jac;  1 -> raw f32 output.
__global__ void k_gemm_bf16(const unsigned short* __restrict__ A,
                            const unsigned short* __restrict__ Bm,
                            const float* __restrict__ bias,
                            int M, int N, int K, int klimMode, int epi,
                            unsigned short* __restrict__ outAct,
                            float* __restrict__ outT,
                            float* __restrict__ outY) {
  int gw   = (blockIdx.x * blockDim.x + threadIdx.x) >> 5;
  int lane = threadIdx.x & 31;
  int tilesN2 = N >> 5;
  int totalW  = (M >> 5) * tilesN2;
  if (gw >= totalW) return;
  int tm = gw / tilesN2, tn = gw - tm * tilesN2;
  int m0 = tm << 5, n0 = tn << 5;

  int klim = K;
  if (klimMode == 1) klim = ((n0 >> 5) + 1) << 5;
  else if (klimMode == 2) { klim = (n0 + 32) << 5; if (klim > K) klim = K; }

  int half = lane >> 4, l15 = lane & 15;
  // A: lane holds row m0(+16)+l15; K runs half*8..+7 and 16+half*8..+7.
  const unsigned short* ap0 = A  + (size_t)(m0 + l15) * K + half * 8;
  const unsigned short* ap1 = A  + (size_t)(m0 + 16 + l15) * K + half * 8;
  // B: lane holds weight row n0(+16)+l15; K runs half*8..+7 and 16+half*8..+7.
  const unsigned short* bp0 = Bm + (size_t)(n0 + l15) * K + half * 8;
  const unsigned short* bp1 = Bm + (size_t)(n0 + 16 + l15) * K + half * 8;

  v8f acc00 = {}, acc01 = {}, acc10 = {}, acc11 = {};
  for (int k = 0; k < klim; k += 32) {
    BfVec a0, a1, b0, b1;
    a0.u[0] = *(const v4u*)(ap0 + k);
    a0.u[1] = *(const v4u*)(ap0 + k + 16);
    a1.u[0] = *(const v4u*)(ap1 + k);
    a1.u[1] = *(const v4u*)(ap1 + k + 16);
    b0.u[0] = *(const v4u*)(bp0 + k);
    b0.u[1] = *(const v4u*)(bp0 + k + 16);
    b1.u[0] = *(const v4u*)(bp1 + k);
    b1.u[1] = *(const v4u*)(bp1 + k + 16);
    __builtin_prefetch(bp0 + k + 512, 0, 1);   // global_prefetch_b8
    __builtin_prefetch(bp1 + k + 512, 0, 1);
    acc00 = __builtin_amdgcn_wmma_f32_16x16x32_bf16(false, a0.v, false, b0.v,
                                                    (short)0, acc00, false, false);
    acc01 = __builtin_amdgcn_wmma_f32_16x16x32_bf16(false, a0.v, false, b1.v,
                                                    (short)0, acc01, false, false);
    acc10 = __builtin_amdgcn_wmma_f32_16x16x32_bf16(false, a1.v, false, b0.v,
                                                    (short)0, acc10, false, false);
    acc11 = __builtin_amdgcn_wmma_f32_16x16x32_bf16(false, a1.v, false, b1.v,
                                                    (short)0, acc11, false, false);
  }

  const float LOG2 = 0.6931471805599453f;
#pragma unroll
  for (int mt = 0; mt < 2; ++mt) {
#pragma unroll
    for (int nt = 0; nt < 2; ++nt) {
      const v8f acc = mt ? (nt ? acc11 : acc10) : (nt ? acc01 : acc00);
      int n = n0 + nt * 16 + l15;
      float bv = bias[n];
      if (epi == 0) {
#pragma unroll
        for (int v = 0; v < 8; ++v) {
          int m = m0 + mt * 16 + v + (half << 3);  // D layout: VGPR v -> M=v+8*half
          float y  = acc[v] + bv;
          float th = tanhf(y);
          float z  = -2.f * y;                     // softplus(-2y), stable
          float sp = fmaxf(z, 0.f) + log1pf(__expf(-fabsf(z)));
          float ld = -2.f * (y + sp - LOG2);       // log d tanh/dy
          outAct[(size_t)m * N + n] = f2bf(th);
          outT[(size_t)m * N + n]   = ld;
        }
      } else {
#pragma unroll
        for (int v = 0; v < 8; ++v) {
          int m = m0 + mt * 16 + v + (half << 3);
          outY[(size_t)m * N + n] = acc[v] + bv;
        }
      }
    }
  }
}

// ---------------------------------------------------------------------------
// Per-(batch,block) log-det chain: one wave per (b,n); lane = k then j.
//  s1[k]  = ld2[n,k] + t1[b,32n+k]
//  s2[j]  = log(sum_k exp(s1-xs) * E1[n,k,j]) + xs + ys[n,j]
//  s3[j]  = s2[j] + t0[b,32n+j] + ld0[n,j]
//  part[b,n] = logsumexp_j s3[j]
__global__ void k_logdet(const float* __restrict__ t0, const float* __restrict__ t1,
                         const float* __restrict__ ld0, const float* __restrict__ ld2,
                         const float* __restrict__ ys, const float* __restrict__ E1,
                         float* __restrict__ part) {
  int gw = (blockIdx.x * blockDim.x + threadIdx.x) >> 5;
  int lane = threadIdx.x & 31;
  if (gw >= BATCH * NB) return;
  int b = gw >> 7, n = gw & (NB - 1);

  float s1 = ld2[n * 32 + lane] + t1[(size_t)b * HF + n * 32 + lane];
  float xs = wave_max(s1);
  float v  = __expf(s1 - xs);

  float accj = 0.f;
  const float* e = E1 + (size_t)n * 1024 + lane;
#pragma unroll 8
  for (int k = 0; k < 32; ++k) {
    float vk = __shfl(v, k, 32);
    accj = fmaf(vk, e[k * 32], accj);
  }
  float s3 = __logf(accj) + xs + ys[n * 32 + lane]
           + t0[(size_t)b * HF + n * 32 + lane] + ld0[n * 32 + lane];
  float m3 = wave_max(s3);
  float es = wave_sum(__expf(s3 - m3));
  if (lane == 0) part[b * NB + n] = m3 + __logf(es);
}

__global__ void k_logdet_reduce(const float* __restrict__ part,
                                float* __restrict__ out) {
  int b = blockIdx.x * (blockDim.x >> 5) + (threadIdx.x >> 5);
  int lane = threadIdx.x & 31;
  if (b >= BATCH) return;
  float s = 0.f;
#pragma unroll
  for (int n = lane; n < NB; n += 32) s += part[b * NB + n];
  s = wave_sum(s);
  if (lane == 0) out[b] = s;
}

// ---------------------------------------------------------------------------
extern "C" void kernel_launch(void* const* d_in, const int* in_sizes, int n_in,
                              void* d_out, int out_size, void* d_ws, size_t ws_size,
                              hipStream_t stream) {
  (void)in_sizes; (void)n_in; (void)out_size; (void)ws_size;
  const float* x   = (const float*)d_in[0];
  const float* W0  = (const float*)d_in[1];
  const float* b0  = (const float*)d_in[2];
  const float* ls0 = (const float*)d_in[3];
  const float* W1  = (const float*)d_in[4];
  const float* b1  = (const float*)d_in[5];
  const float* ls1 = (const float*)d_in[6];
  const float* W2  = (const float*)d_in[7];
  const float* b2  = (const float*)d_in[8];
  const float* ls2 = (const float*)d_in[9];

  char* ws = (char*)d_ws;
  size_t off = 0;
  auto take = [&](size_t bytes) { void* p = ws + off;
                                  off = (off + bytes + 255) & ~(size_t)255; return p; };
  unsigned short* Wn1 = (unsigned short*)take((size_t)HF * HF * 2);   // 32 MB triangle
  unsigned short* Wn0 = (unsigned short*)take((size_t)HF * NB * 2);
  unsigned short* Wn2 = (unsigned short*)take((size_t)NB * HF * 2);
  unsigned short* xbf = (unsigned short*)take((size_t)BATCH * NB * 2);
  unsigned short* a0  = (unsigned short*)take((size_t)BATCH * HF * 2);
  unsigned short* a1  = (unsigned short*)take((size_t)BATCH * HF * 2);
  float* t0  = (float*)take((size_t)BATCH * HF * 4);
  float* t1  = (float*)take((size_t)BATCH * HF * 4);
  float* ld0 = (float*)take((size_t)HF * 4);
  float* ld1 = (float*)take((size_t)NB * 32 * 32 * 4);
  float* ld2 = (float*)take((size_t)NB * 32 * 4);
  float* ys  = (float*)take((size_t)NB * 32 * 4);
  float* E1  = (float*)take((size_t)NB * 32 * 32 * 4);
  float* prt = (float*)take((size_t)BATCH * NB * 4);

  float* outY = (float*)d_out;                    // (256,128)
  float* outL = outY + (size_t)BATCH * NB;        // (256,)

  // prep
  k_cvt_bf16<<<(BATCH * NB + 255) / 256, 256, 0, stream>>>(x, xbf, BATCH * NB);
  k_prep0<<<HF / 8, 256, 0, stream>>>(W0, ls0, Wn0, ld0);
  k_prep1<<<HF / 8, 256, 0, stream>>>(W1, ls1, Wn1, ld1);
  k_prep2<<<NB / 8, 256, 0, stream>>>(W2, ls2, Wn2, ld2);
  k_prepE1<<<(NB * 32) / 256, 256, 0, stream>>>(ld1, ys, E1);

  // layer 0: (256,128) @ (4096,128)^T ; full K (Wn0 zero-padded)
  {
    int waves = (BATCH / 32) * (HF / 32);                // 1024
    k_gemm_bf16<<<(waves * 32 + 255) / 256, 256, 0, stream>>>(
        xbf, Wn0, b0, BATCH, HF, NB, 0, 0, a0, t0, nullptr);
  }
  // layer 1: (256,4096) @ (4096,4096)^T ; triangular K limit
  {
    int waves = (BATCH / 32) * (HF / 32);                // 1024
    k_gemm_bf16<<<(waves * 32 + 255) / 256, 256, 0, stream>>>(
        a0, Wn1, b1, BATCH, HF, HF, 1, 0, a1, t1, nullptr);
  }
  // layer 2: (256,4096) @ (128,4096)^T ; padded-triangular K limit
  {
    int waves = (BATCH / 32) * (NB / 32);                // 32
    k_gemm_bf16<<<(waves * 32 + 255) / 256, 256, 0, stream>>>(
        a1, Wn2, b2, BATCH, NB, HF, 2, 1, nullptr, nullptr, outY);
  }
  // log-det composition
  k_logdet<<<(BATCH * NB) / 8, 256, 0, stream>>>(t0, t1, ld0, ld2, ys, E1, prt);
  k_logdet_reduce<<<BATCH / 8, 256, 0, stream>>>(prt, outL);
}
